// SparseEncoder_AUX_41781441855913
// MI455X (gfx1250) — matrix operand order repaired and checked by reading, not compiled, not run on hardware
//
#include <hip/hip_runtime.h>
#include <hip/hip_bf16.h>

#define EPSV 0.001f
#define SLOPEV 0.01f
#define BIGV 1e10f
#define NPTS 16384

typedef __attribute__((ext_vector_type(16))) _Float16 v16h;
typedef __attribute__((ext_vector_type(8)))  float    v8f;
typedef __attribute__((ext_vector_type(4)))  unsigned v4u;

union Frag16 { v16h v; v4u q[2]; };

// ---------------------------------------------------------------------------
// utility kernels
// ---------------------------------------------------------------------------
__global__ void k_zero(float* __restrict__ p, long n) {
    long i = (long)blockIdx.x * blockDim.x + threadIdx.x;
    if (i < n) p[i] = 0.f;
}

__global__ void k_scatter(const float* __restrict__ vf, const int* __restrict__ coors,
                          _Float16* __restrict__ act0, float* __restrict__ mask0,
                          int N, int H, int W, int Cpad) {
    int i = blockIdx.x * blockDim.x + threadIdx.x;
    if (i >= N) return;
    int z = coors[i * 4 + 1], yy = coors[i * 4 + 2], xx = coors[i * 4 + 3];
    long v = ((long)z * H + yy) * W + xx;
    mask0[v] = 1.f;
    for (int c = 0; c < 4; ++c) act0[v * Cpad + c] = (_Float16)vf[i * 4 + c];
}

// repack DHWIO f32 weights -> [tap][cout][cin_padded] f16 (B-matrix friendly)
__global__ void k_repack_w(const float* __restrict__ w, _Float16* __restrict__ wt,
                           int taps, int Ci, int CiPad, int Co) {
    long i = (long)blockIdx.x * blockDim.x + threadIdx.x;
    long total = (long)taps * Co * CiPad;
    if (i >= total) return;
    int ci = (int)(i % CiPad);
    long t = i / CiPad;
    int co = (int)(t % Co);
    int tap = (int)(t / Co);
    float val = (ci < Ci) ? w[((long)tap * Ci + ci) * Co + co] : 0.f;
    wt[i] = (_Float16)val;
}

// downsampled mask: conv(mask, ones) > 0
__global__ void k_mask_down(const float* __restrict__ mIn, float* __restrict__ mOut,
                            int iD, int iH, int iW, int oD, int oH, int oW,
                            int kd, int kh, int kw, int sz, int sy, int sx,
                            int pz, int py, int px) {
    long v = (long)blockIdx.x * blockDim.x + threadIdx.x;
    long V = (long)oD * oH * oW;
    if (v >= V) return;
    int ox = (int)(v % oW); long t = v / oW;
    int oy = (int)(t % oH); int oz = (int)(t / oH);
    float s = 0.f;
    for (int dz = 0; dz < kd; ++dz)
        for (int dy = 0; dy < kh; ++dy)
            for (int dx = 0; dx < kw; ++dx) {
                int zz = oz * sz + dz - pz, yy = oy * sy + dy - py, xx = ox * sx + dx - px;
                if ((unsigned)zz < (unsigned)iD && (unsigned)yy < (unsigned)iH &&
                    (unsigned)xx < (unsigned)iW)
                    s += mIn[((long)zz * iH + yy) * iW + xx];
            }
    mOut[v] = (s > 0.f) ? 1.f : 0.f;
}

// ---------------------------------------------------------------------------
// implicit-GEMM 3D conv via v_wmma_f32_16x16x32_f16, LDS-staged A tile.
// Templated on compile-time shape so the unrolled K loop folds tap decode and
// LDS/weight offsets into load immediates: per chunk 2x ds_load_b128 (A) +
// 2x global_load_b128 (B) + one WMMA. kd is always 3 in this network.
// ---------------------------------------------------------------------------
extern __shared__ unsigned smemU[];

template <int CIN, int COUT, int KW, int KH, int SX>
__global__ void k_conv_wmma(const _Float16* __restrict__ actIn,
                            const _Float16* __restrict__ wt,
                            const float* __restrict__ maskOut,
                            float* __restrict__ yOut,
                            int iD, int iH, int iW, int oD, int oH, int oW,
                            int sz, int sy, int pz, int py, int px, int xTiles) {
    constexpr int KD = 3;
    constexpr int ROWS = KD * KH;
    constexpr int WLDS = 15 * SX + KW;
    constexpr int CHALF = CIN / 2;
    constexpr int CMASK = CIN - 1;
    constexpr int CSH = (CIN == 16) ? 4 : (CIN == 32) ? 5 : 6;
    constexpr int TAPS = KD * KH * KW;
    constexpr int KK = TAPS * CIN;
    constexpr int NCH = (KK + 31) / 32;

    const int xTile = blockIdx.x % xTiles;
    const int rowId = blockIdx.x / xTiles;
    const int oy = rowId % oH;
    const int oz = rowId / oH;
    const int x0 = xTile * 16;
    const int tid = threadIdx.x;
    const int lane = tid & 31;
    const int n0 = (tid >> 5) * 16;            // wave id -> cout tile
    const int m = lane & 15;
    const int hh = lane >> 4;
    const int xBase = x0 * SX - px;
    const v4u ZERO4 = {0u, 0u, 0u, 0u};

    // pull this wave's weight tile toward the caches (global_prefetch_b8)
    __builtin_prefetch(wt + (long)n0 * CIN, 0, 3);

    // ---- stage receptive field into LDS: [row(KD*KH)][WLDS][CIN] f16 ----
    constexpr int TOTALDW = ROWS * WLDS * CHALF;
    for (int q = tid; q < TOTALDW; q += blockDim.x) {
        int ci2 = q % CHALF;
        int t = q / CHALF;
        int xi = t % WLDS;
        int r = t / WLDS;
        int dz = r / KH, dy = r % KH;
        int zz = oz * sz + dz - pz;
        int yy = oy * sy + dy - py;
        int xx = xBase + xi;
        unsigned val = 0u;
        if ((unsigned)zz < (unsigned)iD && (unsigned)yy < (unsigned)iH &&
            (unsigned)xx < (unsigned)iW) {
            long nb = ((long)zz * iH + yy) * iW + xx;
            val = *(const unsigned*)(actIn + nb * CIN + 2 * ci2);
        }
        smemU[q] = val;
    }
    __syncthreads();

    v8f acc = {0.f, 0.f, 0.f, 0.f, 0.f, 0.f, 0.f, 0.f};
    const int n = m;
    const int mBase = m * SX * CHALF;                       // lane's LDS column base
    const _Float16* wBase = wt + (long)(n0 + n) * CIN;      // lane's weight column base

#pragma unroll
    for (int ch = 0; ch < NCH; ++ch) {
        const int kBase = ch * 32;
        // each 16-half of a chunk lies in one tap (CIN multiple of 16)
        const int tapLo = kBase >> CSH;
        const bool hiValid = (kBase + 16) < KK;
        const int tapHi = hiValid ? ((kBase + 16) >> CSH) : 0;
        const int rLo = tapLo / KW, dxLo = tapLo % KW;
        const int rHi = tapHi / KW, dxHi = tapHi % KW;
        const int axLo = (rLo * WLDS + dxLo) * CHALF + mBase;
        const int axHi = (rHi * WLDS + dxHi) * CHALF + mBase;

        // A fragment (16x32 f16): two 16B LDS loads per lane
        Frag16 fa;
        {
            int ci2 = ((kBase + hh * 8) & CMASK) >> 1;
            fa.q[0] = *(const v4u*)(smemU + axLo + ci2);
            if (hiValid) {
                int ci2h = ((kBase + 16 + hh * 8) & CMASK) >> 1;
                fa.q[1] = *(const v4u*)(smemU + axHi + ci2h);
            } else {
                fa.q[1] = ZERO4;
            }
        }
        // B fragment (32x16 f16): two 16B global loads per lane
        Frag16 fb;
        if (hiValid) {
            fb.q[0] = *(const v4u*)(wBase + (long)tapLo * (COUT * CIN) +
                                    ((kBase + hh * 8) & CMASK));
            fb.q[1] = *(const v4u*)(wBase + (long)tapHi * (COUT * CIN) +
                                    ((kBase + 16 + hh * 8) & CMASK));
        } else {
            fb.q[0] = *(const v4u*)(wBase + (long)tapLo * (COUT * CIN) +
                                    ((kBase + hh * 8) & CMASK));
            fb.q[1] = ZERO4;
        }
        acc = __builtin_amdgcn_wmma_f32_16x16x32_f16(false, fa.v, false, fb.v, (short)0,
                                                     acc, false, false);
    }
    // C/D layout: VGPR r -> M = r + 8*half, N = lane&15
#pragma unroll
    for (int r = 0; r < 8; ++r) {
        int x = x0 + r + 8 * hh;
        if (x < oW) {
            long vout = (long)rowId * oW + x;
            float mk = maskOut[vout];
            yOut[vout * COUT + n0 + n] = acc[r] * mk;
        }
    }
}

template <int CIN, int COUT, int KW, int KH, int SX>
static void launchConv(const _Float16* in, const _Float16* wtb, const float* mOut,
                       float* y, int iD, int iH, int iW, int oD, int oH, int oW,
                       int sz, int sy, int pz, int py, int px, hipStream_t stream) {
    int xTiles = (oW + 15) / 16;
    int numRows = oD * oH;
    constexpr int WLDS = 15 * SX + KW;
    size_t shBytes = (size_t)3 * KH * WLDS * CIN * 2;
    k_conv_wmma<CIN, COUT, KW, KH, SX>
        <<<numRows * xTiles, 32 * (COUT / 16), shBytes, stream>>>(
            in, wtb, mOut, y, iD, iH, iW, oD, oH, oW, sz, sy, pz, py, px, xTiles);
}

// ---------------------------------------------------------------------------
// masked BN statistics (per-channel sum, sumsq; mask count at [2C])
// ---------------------------------------------------------------------------
__global__ void k_stats(const float* __restrict__ y, const float* __restrict__ mask,
                        float* __restrict__ stats, long V, int C) {
    long gtid = (long)blockIdx.x * blockDim.x + threadIdx.x;
    int c = (int)(gtid % C);
    long v = gtid / C;
    long stride = ((long)gridDim.x * blockDim.x) / C;
    float s = 0.f, q = 0.f, mcnt = 0.f;
    for (; v < V; v += stride) {
        float val = y[v * C + c];
        s += val;
        q += val * val;
        if (c == 0) mcnt += mask[v];
    }
    atomicAdd(&stats[c], s);
    atomicAdd(&stats[C + c], q);
    if (c == 0) atomicAdd(&stats[2 * C], mcnt);
}

__global__ void k_bnact(const float* __restrict__ y, const float* __restrict__ mask,
                        const float* __restrict__ stats, const float* __restrict__ g,
                        const float* __restrict__ b, _Float16* __restrict__ actOut,
                        float* __restrict__ hOut, long V, int C) {
    long i = (long)blockIdx.x * blockDim.x + threadIdx.x;
    if (i >= V * C) return;
    int c = (int)(i % C);
    long v = i / C;
    float cnt = fmaxf(stats[2 * C], 1.f);
    float mu = stats[c] / cnt;
    float var = stats[C + c] / cnt - mu * mu;
    float t = (y[i] - mu) * rsqrtf(var + EPSV) * g[c] + b[c];
    t = (t >= 0.f) ? t : SLOPEV * t;
    t *= mask[v];
    actOut[i] = (_Float16)t;
    if (hOut) hOut[i] = t;
}

// ---------------------------------------------------------------------------
// brute-force 3-NN over grid centers with invalid-voxel penalty
// ---------------------------------------------------------------------------
__global__ void k_knn(const float* __restrict__ vf, const float* __restrict__ mask,
                      float* __restrict__ rec, int* __restrict__ idx, int N,
                      int D, int H, int W, float vx, float vy, float vz) {
    int p = blockIdx.x * blockDim.x + threadIdx.x;
    if (p >= N) return;
    float px = vf[p * 4 + 0], py = vf[p * 4 + 1], pz = vf[p * 4 + 2];
    float b0 = 1e30f, b1 = 1e30f, b2 = 1e30f;
    int i0 = 0, i1 = 0, i2 = 0;
    long v = 0;
    for (int z = 0; z < D; ++z) {
        float cz = ((float)z + 0.5f) * vz - 3.0f;
        float dz = pz - cz;
        float dz2 = dz * dz;
        for (int y = 0; y < H; ++y) {
            float cy = ((float)y + 0.5f) * vy - 40.0f;
            float dy = py - cy;
            float dyz = dz2 + dy * dy;
            for (int x = 0; x < W; ++x, ++v) {
                float cx = ((float)x + 0.5f) * vx;
                float dx = px - cx;
                float d2 = dyz + dx * dx + (1.0f - mask[v]) * BIGV;
                if (d2 < b0) { b2 = b1; i2 = i1; b1 = b0; i1 = i0; b0 = d2; i0 = (int)v; }
                else if (d2 < b1) { b2 = b1; i2 = i1; b1 = d2; i1 = (int)v; }
                else if (d2 < b2) { b2 = d2; i2 = (int)v; }
            }
        }
    }
    float d;
    d = sqrtf(fmaxf(b0, 0.f)); rec[p * 3 + 0] = 1.f / (d + 1e-8f); idx[p * 3 + 0] = i0;
    d = sqrtf(fmaxf(b1, 0.f)); rec[p * 3 + 1] = 1.f / (d + 1e-8f); idx[p * 3 + 1] = i1;
    d = sqrtf(fmaxf(b2, 0.f)); rec[p * 3 + 2] = 1.f / (d + 1e-8f); idx[p * 3 + 2] = i2;
}

// reference quirk: rec.sum(axis=0) sums over POINTS per neighbor slot
__global__ void k_recsum(const float* __restrict__ rec, float* __restrict__ norm, int N) {
    int p = blockIdx.x * blockDim.x + threadIdx.x;
    if (p >= N) return;
    atomicAdd(&norm[0], rec[p * 3 + 0]);
    atomicAdd(&norm[1], rec[p * 3 + 1]);
    atomicAdd(&norm[2], rec[p * 3 + 2]);
}

__global__ void k_interp(const float* __restrict__ h, const float* __restrict__ rec,
                         const float* __restrict__ norm, const int* __restrict__ idx,
                         _Float16* __restrict__ pcat, int N, int C, int colOff,
                         int colStride) {
    long i = (long)blockIdx.x * blockDim.x + threadIdx.x;
    if (i >= (long)N * C) return;
    int c = (int)(i % C);
    int p = (int)(i / C);
    float acc = 0.f;
#pragma unroll
    for (int j = 0; j < 3; ++j) {
        float w = rec[p * 3 + j] / norm[j];
        acc += h[(long)idx[p * 3 + j] * C + c] * w;
    }
    pcat[(long)p * colStride + colOff + c] = (_Float16)acc;
}

// ---------------------------------------------------------------------------
// FC (16384x160 @ 160x64) via WMMA; Bt is f16 [co][k]; 16B fragment loads
// ---------------------------------------------------------------------------
__global__ __launch_bounds__(32) void k_fc_wmma(const _Float16* __restrict__ A,
                                                const _Float16* __restrict__ Bt,
                                                float* __restrict__ out, int Nrows,
                                                int K, int Co, int nTiles) {
    const int wave = blockIdx.x;
    const int mTile = wave / nTiles;
    const int n0 = (wave - mTile * nTiles) * 16;
    const int lane = threadIdx.x;
    const int m = lane & 15;
    const int hh = lane >> 4;
    const long row = (long)mTile * 16 + m;
    v8f acc = {0.f, 0.f, 0.f, 0.f, 0.f, 0.f, 0.f, 0.f};
    for (int kBase = 0; kBase < K; kBase += 32) {
        Frag16 fa, fb;
        const _Float16* ap = A + row * K + kBase + hh * 8;
        fa.q[0] = *(const v4u*)(ap);
        fa.q[1] = *(const v4u*)(ap + 16);
        const _Float16* bp = Bt + (long)(n0 + m) * K + kBase + hh * 8;
        fb.q[0] = *(const v4u*)(bp);
        fb.q[1] = *(const v4u*)(bp + 16);
        acc = __builtin_amdgcn_wmma_f32_16x16x32_f16(false, fa.v, false, fb.v, (short)0,
                                                     acc, false, false);
    }
#pragma unroll
    for (int r = 0; r < 8; ++r) {
        long ro = (long)mTile * 16 + r + 8 * hh;
        out[ro * Co + n0 + m] = acc[r];
    }
}

__global__ void k_heads(const float* __restrict__ pw, const float* __restrict__ clsW,
                        const float* __restrict__ regW, const float* __restrict__ vf,
                        const int* __restrict__ coors, float* __restrict__ pm,
                        float* __restrict__ ocls, float* __restrict__ oreg, int N) {
    int p = blockIdx.x * blockDim.x + threadIdx.x;
    if (p >= N) return;
    float c0 = 0.f, r0 = 0.f, r1 = 0.f, r2 = 0.f;
    for (int k = 0; k < 64; ++k) {
        float v = pw[(long)p * 64 + k];
        c0 += v * clsW[k];
        r0 += v * regW[k * 3 + 0];
        r1 += v * regW[k * 3 + 1];
        r2 += v * regW[k * 3 + 2];
    }
    ocls[p] = c0;
    oreg[p * 3 + 0] = r0; oreg[p * 3 + 1] = r1; oreg[p * 3 + 2] = r2;
    pm[p * 4 + 0] = (float)coors[p * 4 + 0];
    pm[p * 4 + 1] = vf[p * 4 + 0];
    pm[p * 4 + 2] = vf[p * 4 + 1];
    pm[p * 4 + 3] = vf[p * 4 + 2];
}

// xv = transpose of extra-layer output: xv[c][y][x] = h[(y*W+x)*C + c]
__global__ void k_xv(const float* __restrict__ h, float* __restrict__ xv, int HH, int WW,
                     int C) {
    int i = blockIdx.x * blockDim.x + threadIdx.x;
    int total = C * HH * WW;
    if (i >= total) return;
    int x = i % WW;
    int t = i / WW;
    int y = t % HH;
    int c = t / HH;
    xv[i] = h[((long)y * WW + x) * C + c];
}

// ---------------------------------------------------------------------------
// host orchestration
// ---------------------------------------------------------------------------
static inline size_t alignUp(size_t x, size_t a) { return (x + a - 1) & ~(a - 1); }

extern "C" void kernel_launch(void* const* d_in, const int* in_sizes, int n_in,
                              void* d_out, int out_size, void* d_ws, size_t ws_size,
                              hipStream_t stream) {
    (void)in_sizes; (void)n_in; (void)out_size; (void)ws_size;

    const int N = NPTS;
    // level dims
    const int D0 = 20, Hd0 = 200, Wd0 = 176;
    const int D1 = 10, Hd1 = 100, Wd1 = 88;
    const int D2 = 5,  Hd2 = 50,  Wd2 = 44;
    const int D3 = 3,  Hd3 = 25,  Wd3 = 22;
    const int DX = 1,  HdX = 25,  WdX = 22;
    const long V0 = (long)D0 * Hd0 * Wd0;
    const long V1 = (long)D1 * Hd1 * Wd1;
    const long V2 = (long)D2 * Hd2 * Wd2;
    const long V3 = (long)D3 * Hd3 * Wd3;
    const long VX = (long)DX * HdX * WdX;

    // inputs (setup_inputs dict flattening order)
    const float* vfeat = (const float*)d_in[0];
    struct LP { const float *w, *g, *b; };
    auto getLP = [&](int base) -> LP {
        return {(const float*)d_in[base], (const float*)d_in[base + 1],
                (const float*)d_in[base + 2]};
    };
    LP c00 = getLP(1), c01 = getLP(4), dn0 = getLP(7);
    LP c10 = getLP(10), c11 = getLP(13), dn1 = getLP(16);
    LP c20 = getLP(19), c21 = getLP(22), c22 = getLP(25), dn2 = getLP(28);
    LP c30 = getLP(31), c31 = getLP(34), c32 = getLP(37), ex = getLP(40);
    const float* fcW  = (const float*)d_in[43];
    const float* clsW = (const float*)d_in[44];
    const float* regW = (const float*)d_in[45];
    const int* coors  = (const int*)d_in[46];

    // workspace layout
    char* ws = (char*)d_ws;
    size_t off = 0;
    auto alloc = [&](size_t bytes) -> char* {
        char* p = ws + off;
        off = alignUp(off + bytes, 256);
        return p;
    };
    float* mask0 = (float*)alloc(V0 * 4);
    float* mask1 = (float*)alloc(V1 * 4);
    float* mask2 = (float*)alloc(V2 * 4);
    float* mask3 = (float*)alloc(V3 * 4);
    float* maskX = (float*)alloc(VX * 4);
    _Float16* act0a = (_Float16*)alloc(V0 * 16 * 2);  // region A (reused below)
    _Float16* act0b = (_Float16*)alloc(V0 * 16 * 2);
    float* y0 = (float*)alloc(V0 * 16 * 4);           // region Y (reused below)
    // sub-buffers packed inside dead regions (liveness checked):
    _Float16* act1a = act0a;
    _Float16* act1b = act0a + V1 * 32;
    _Float16* act2a = act0a + 2 * V1 * 32;
    _Float16* act2b = act2a + V2 * 64;
    _Float16* act3a = act2b + V2 * 64;
    _Float16* act3b = act3a + V3 * 64;
    float* y1 = y0;
    float* h1 = y0 + V1 * 32;
    float* y2 = h1 + V1 * 32;
    float* h2 = y2 + V2 * 64;
    float* y3 = h2 + V2 * 64;
    float* h3 = y3 + V3 * 64;
    float* yx = h3 + V3 * 64;
    float* hx = yx + VX * 64;
    _Float16* wt = (_Float16*)alloc((size_t)27 * 64 * 64 * 2);
    float* stats = (float*)alloc(129 * 4);
    float* rec = (float*)alloc((size_t)N * 3 * 4);
    int* idxb = (int*)alloc((size_t)N * 3 * 4);
    float* norm = (float*)alloc(16);
    _Float16* pcat = (_Float16*)alloc((size_t)N * 160 * 2);
    float* pw = (float*)alloc((size_t)N * 64 * 4);

    float* out = (float*)d_out;
    float* oXV = out;                  // 64*25*22 = 35200
    float* oPM = out + 35200;          // 16384*4
    float* oCLS = out + 100736;        // 16384
    float* oREG = out + 117120;        // 16384*3

    auto zero32 = [&](float* p, long n) {
        k_zero<<<(int)((n + 255) / 256), 256, 0, stream>>>(p, n);
    };
    auto repack = [&](const float* w, int taps, int Ci, int CiPad, int Co) {
        long wn = (long)taps * Co * CiPad;
        k_repack_w<<<(int)((wn + 255) / 256), 256, 0, stream>>>(w, wt, taps, Ci, CiPad,
                                                                Co);
    };
    auto runBN = [&](float* y, const float* m, const float* g, const float* b,
                     _Float16* actOut, float* hOut, long V, int C) {
        zero32(stats, 129);
        k_stats<<<512, 256, 0, stream>>>(y, m, stats, V, C);
        long n = V * (long)C;
        k_bnact<<<(int)((n + 255) / 256), 256, 0, stream>>>(y, m, stats, g, b, actOut,
                                                            hOut, V, C);
    };
    auto runInterp = [&](const float* h, const float* m, int D, int H, int W, float vx,
                         float vy, float vz, int C, int colOff) {
        k_knn<<<(N + 127) / 128, 128, 0, stream>>>(vfeat, m, rec, idxb, N, D, H, W, vx,
                                                   vy, vz);
        zero32(norm, 4);
        k_recsum<<<(N + 255) / 256, 256, 0, stream>>>(rec, norm, N);
        long n = (long)N * C;
        k_interp<<<(int)((n + 255) / 256), 256, 0, stream>>>(h, rec, norm, idxb, pcat, N,
                                                             C, colOff, 160);
    };

    // ---- scatter input voxels ----
    zero32((float*)act0a, V0 * 8);   // V0*16 f16 == V0*8 f32
    zero32(mask0, V0);
    k_scatter<<<(N + 255) / 256, 256, 0, stream>>>(vfeat, coors, act0a, mask0, N, Hd0,
                                                   Wd0, 16);

    // ---- level 0: conv0 (4->16, 16->16) ----
    repack(c00.w, 27, 4, 16, 16);
    launchConv<16, 16, 3, 3, 1>(act0a, wt, mask0, y0, D0, Hd0, Wd0, D0, Hd0, Wd0,
                                1, 1, 1, 1, 1, stream);
    runBN(y0, mask0, c00.g, c00.b, act0b, nullptr, V0, 16);
    repack(c01.w, 27, 16, 16, 16);
    launchConv<16, 16, 3, 3, 1>(act0b, wt, mask0, y0, D0, Hd0, Wd0, D0, Hd0, Wd0,
                                1, 1, 1, 1, 1, stream);
    runBN(y0, mask0, c01.g, c01.b, act0a, nullptr, V0, 16);

    // ---- down0 (16->32, stride 2) ----
    k_mask_down<<<(int)((V1 + 255) / 256), 256, 0, stream>>>(
        mask0, mask1, D0, Hd0, Wd0, D1, Hd1, Wd1, 3, 3, 3, 2, 2, 2, 1, 1, 1);
    repack(dn0.w, 27, 16, 16, 32);
    launchConv<16, 32, 3, 3, 2>(act0a, wt, mask1, y1, D0, Hd0, Wd0, D1, Hd1, Wd1,
                                2, 2, 1, 1, 1, stream);
    runBN(y1, mask1, dn0.g, dn0.b, act1a, nullptr, V1, 32);

    // ---- level 1: conv1 (32->32)x2 ----
    repack(c10.w, 27, 32, 32, 32);
    launchConv<32, 32, 3, 3, 1>(act1a, wt, mask1, y1, D1, Hd1, Wd1, D1, Hd1, Wd1,
                                1, 1, 1, 1, 1, stream);
    runBN(y1, mask1, c10.g, c10.b, act1b, nullptr, V1, 32);
    repack(c11.w, 27, 32, 32, 32);
    launchConv<32, 32, 3, 3, 1>(act1b, wt, mask1, y1, D1, Hd1, Wd1, D1, Hd1, Wd1,
                                1, 1, 1, 1, 1, stream);
    runBN(y1, mask1, c11.g, c11.b, act1a, h1, V1, 32);
    runInterp(h1, mask1, D1, Hd1, Wd1, 0.1f, 0.1f, 0.2f, 32, 0);

    // ---- down1 (32->64) ----
    k_mask_down<<<(int)((V2 + 255) / 256), 256, 0, stream>>>(
        mask1, mask2, D1, Hd1, Wd1, D2, Hd2, Wd2, 3, 3, 3, 2, 2, 2, 1, 1, 1);
    repack(dn1.w, 27, 32, 32, 64);
    launchConv<32, 64, 3, 3, 2>(act1a, wt, mask2, y2, D1, Hd1, Wd1, D2, Hd2, Wd2,
                                2, 2, 1, 1, 1, stream);
    runBN(y2, mask2, dn1.g, dn1.b, act2a, nullptr, V2, 64);

    // ---- level 2: conv2 (64->64)x3 ----
    repack(c20.w, 27, 64, 64, 64);
    launchConv<64, 64, 3, 3, 1>(act2a, wt, mask2, y2, D2, Hd2, Wd2, D2, Hd2, Wd2,
                                1, 1, 1, 1, 1, stream);
    runBN(y2, mask2, c20.g, c20.b, act2b, nullptr, V2, 64);
    repack(c21.w, 27, 64, 64, 64);
    launchConv<64, 64, 3, 3, 1>(act2b, wt, mask2, y2, D2, Hd2, Wd2, D2, Hd2, Wd2,
                                1, 1, 1, 1, 1, stream);
    runBN(y2, mask2, c21.g, c21.b, act2a, nullptr, V2, 64);
    repack(c22.w, 27, 64, 64, 64);
    launchConv<64, 64, 3, 3, 1>(act2a, wt, mask2, y2, D2, Hd2, Wd2, D2, Hd2, Wd2,
                                1, 1, 1, 1, 1, stream);
    runBN(y2, mask2, c22.g, c22.b, act2b, h2, V2, 64);
    runInterp(h2, mask2, D2, Hd2, Wd2, 0.2f, 0.2f, 0.4f, 64, 32);

    // ---- down2 (64->64) ----
    k_mask_down<<<(int)((V3 + 255) / 256), 256, 0, stream>>>(
        mask2, mask3, D2, Hd2, Wd2, D3, Hd3, Wd3, 3, 3, 3, 2, 2, 2, 1, 1, 1);
    repack(dn2.w, 27, 64, 64, 64);
    launchConv<64, 64, 3, 3, 2>(act2b, wt, mask3, y3, D2, Hd2, Wd2, D3, Hd3, Wd3,
                                2, 2, 1, 1, 1, stream);
    runBN(y3, mask3, dn2.g, dn2.b, act3a, nullptr, V3, 64);

    // ---- level 3: conv3 (64->64)x3 ----
    repack(c30.w, 27, 64, 64, 64);
    launchConv<64, 64, 3, 3, 1>(act3a, wt, mask3, y3, D3, Hd3, Wd3, D3, Hd3, Wd3,
                                1, 1, 1, 1, 1, stream);
    runBN(y3, mask3, c30.g, c30.b, act3b, nullptr, V3, 64);
    repack(c31.w, 27, 64, 64, 64);
    launchConv<64, 64, 3, 3, 1>(act3b, wt, mask3, y3, D3, Hd3, Wd3, D3, Hd3, Wd3,
                                1, 1, 1, 1, 1, stream);
    runBN(y3, mask3, c31.g, c31.b, act3a, nullptr, V3, 64);
    repack(c32.w, 27, 64, 64, 64);
    launchConv<64, 64, 3, 3, 1>(act3a, wt, mask3, y3, D3, Hd3, Wd3, D3, Hd3, Wd3,
                                1, 1, 1, 1, 1, stream);
    runBN(y3, mask3, c32.g, c32.b, act3b, h3, V3, 64);
    runInterp(h3, mask3, D3, Hd3, Wd3, 0.4f, 0.4f, 0.8f, 64, 96);

    // ---- extra (64->64, kernel (3,1,1), stride (2,1,1), pad 0) ----
    k_mask_down<<<(int)((VX + 255) / 256), 256, 0, stream>>>(
        mask3, maskX, D3, Hd3, Wd3, DX, HdX, WdX, 3, 1, 1, 2, 1, 1, 0, 0, 0);
    repack(ex.w, 3, 64, 64, 64);
    launchConv<64, 64, 1, 1, 1>(act3b, wt, maskX, yx, D3, Hd3, Wd3, DX, HdX, WdX,
                                2, 1, 0, 0, 0, stream);
    runBN(yx, maskX, ex.g, ex.b, act3a /*dummy*/, hx, VX, 64);
    k_xv<<<(35200 + 255) / 256, 256, 0, stream>>>(hx, oXV, HdX, WdX, 64);

    // ---- point head: pw = [p1|p2|p3] @ fc, then cls/reg ----
    {
        repack(fcW, 1, 160, 160, 64);
        int nTiles = 64 / 16;
        k_fc_wmma<<<(N / 16) * nTiles, 32, 0, stream>>>(pcat, wt, pw, N, 160, 64, nTiles);
        k_heads<<<(N + 255) / 256, 256, 0, stream>>>(pw, clsW, regW, vfeat, coors, oPM,
                                                     oCLS, oREG, N);
    }
}
